// EGbackbone_72524817760275
// MI455X (gfx1250) — matrix operand order repaired and checked by reading, compiled, not run on hardware
//
#include <hip/hip_runtime.h>
#include <math.h>

// ---------------------------------------------------------------------------
// EGbackbone forward for gfx1250 (MI455X).
//
// Input pointer layout (recursive dict-insertion-order flattening of
// setup_inputs()):
//   0 feats(8192,1) 1 points0(8192,3) 2..5 neighbors0..3  6..8 subsampling0..2
//   9 upsampling1(2048,1) 10 upsampling2(512,1)
//   11..13 e11{kw,gs,gb}
//   14..28 e12{u1w,u1b,u1gs,u1gb,kw,g2s,g2b,u2w,u2b,g3s,g3b,sw,sb,sgs,sgb}
//   29..39 e21 (no sw)  40..50 e22  51..61 e31  62..72 e32
//   73..87 e41 (sw)     88..102 e42 (sw)
//   103..105 egm1{qw,kw,vw} 106..108 egm2 109..111 egm3
//   112 l1w 113 l1b 114 l2w 115 l2b 116 l3w 117 l3b
//   118..121 d3{uw,ub,ugs,ugb} 122 d2w 123 d2b
// Output: [l2 (2048x64)][l3 (512x128)][f4 (128x1024)] f32, concatenated.
//
// All GEMMs are exactly tiled (M%16==0, N%16==0, K%32==0) by construction:
// kpconv weights are transposed+zero-padded into workspace so no fragment
// load/store needs a bounds check -> pure b128 loads + v_wmma + b32 stores.
// ---------------------------------------------------------------------------

typedef __attribute__((ext_vector_type(16))) _Float16 v16h;
typedef __attribute__((ext_vector_type(8)))  float    v8f;

#define NEG_SLOPE 0.1f
__device__ __forceinline__ float lrelu_f(float x) { return x >= 0.f ? x : NEG_SLOPE * x; }

// ---------------------------------------------------------------------------
// WMMA GEMM: C(MxN) = A(MxK) @ B^T + bias, B in weight layout (N x K).
// Requires M%16==0, N%16==0, K%32==0. One wave per 16x16 C tile,
// v_wmma_f32_16x16x32_f16 per 32-deep K step, unconditional float4 loads.
// ---------------------------------------------------------------------------
__global__ void k_gemm_bias(const float* __restrict__ A, const float* __restrict__ B,
                            const float* __restrict__ bias, float* __restrict__ C,
                            int M, int N, int K)
{
    int lane = threadIdx.x;                 // 0..31
    int wave = threadIdx.y;                 // 0..3
    int mt = blockIdx.y * 4 + wave;
    int nt = blockIdx.x;
    if (mt * 16 >= M) return;               // wave-uniform (never ragged here)
    int half = lane >> 4;
    int lm   = lane & 15;
    const float* arow = A + (long)(mt * 16 + lm) * K;   // A row m, K-contig
    const float* brow = B + (long)(nt * 16 + lm) * K;   // B row n, K-contig
    v8f acc = {};
    for (int kk = 0; kk < K; kk += 32) {
        const float4 a0 = *(const float4*)(arow + kk + half * 8);
        const float4 a1 = *(const float4*)(arow + kk + half * 8 + 4);
        const float4 a2 = *(const float4*)(arow + kk + half * 8 + 16);
        const float4 a3 = *(const float4*)(arow + kk + half * 8 + 20);
        const float4 b0 = *(const float4*)(brow + kk + half * 8);
        const float4 b1 = *(const float4*)(brow + kk + half * 8 + 4);
        const float4 b2 = *(const float4*)(brow + kk + half * 8 + 16);
        const float4 b3 = *(const float4*)(brow + kk + half * 8 + 20);
        v16h a, b;
        a[0]=(_Float16)a0.x; a[1]=(_Float16)a0.y; a[2]=(_Float16)a0.z; a[3]=(_Float16)a0.w;
        a[4]=(_Float16)a1.x; a[5]=(_Float16)a1.y; a[6]=(_Float16)a1.z; a[7]=(_Float16)a1.w;
        a[8]=(_Float16)a2.x; a[9]=(_Float16)a2.y; a[10]=(_Float16)a2.z; a[11]=(_Float16)a2.w;
        a[12]=(_Float16)a3.x; a[13]=(_Float16)a3.y; a[14]=(_Float16)a3.z; a[15]=(_Float16)a3.w;
        b[0]=(_Float16)b0.x; b[1]=(_Float16)b0.y; b[2]=(_Float16)b0.z; b[3]=(_Float16)b0.w;
        b[4]=(_Float16)b1.x; b[5]=(_Float16)b1.y; b[6]=(_Float16)b1.z; b[7]=(_Float16)b1.w;
        b[8]=(_Float16)b2.x; b[9]=(_Float16)b2.y; b[10]=(_Float16)b2.z; b[11]=(_Float16)b2.w;
        b[12]=(_Float16)b3.x; b[13]=(_Float16)b3.y; b[14]=(_Float16)b3.z; b[15]=(_Float16)b3.w;
        acc = __builtin_amdgcn_wmma_f32_16x16x32_f16(false, a, false, b,
                                                     (short)0, acc, false, false);
    }
    float bv = bias ? bias[nt * 16 + lm] : 0.f;
    int n = nt * 16 + lm;
#pragma unroll
    for (int r = 0; r < 8; ++r)
        C[(long)(mt * 16 + half * 8 + r) * N + n] = acc[r] + bv;
}

// Transpose + zero-pad a (Kin x N) row-major weight into (N x Kout) weight
// layout (K-contiguous), rows k >= Kin are zero. Used for kpconv weights.
__global__ void k_pad_transpose(const float* __restrict__ bkn, float* __restrict__ bnk,
                                int N, int Kin, int Kout)
{
    long i = (long)blockIdx.x * blockDim.x + threadIdx.x;
    if (i >= (long)N * Kout) return;
    int n = (int)(i / Kout), k = (int)(i % Kout);
    bnk[i] = (k < Kin) ? bkn[(long)k * N + n] : 0.f;
}

// ---------------------------------------------------------------------------
// KPConv influence + gather + wf = w^T @ f  (per query point, one wave each).
// wf[n,p,c] = sum_k w[n,k,p] * feats[neigh[n,k], c], stored padded to 16 rows
// (row 15 is exact zero since the A-operand row 15 is zeroed in LDS), so the
// downstream GEMM runs with K = 16*Cin and no tails.
// ---------------------------------------------------------------------------
__global__ void k_kpconv_wf(const float* __restrict__ qp, const float* __restrict__ sp,
                            const int* __restrict__ neigh, const float* __restrict__ x,
                            const float* __restrict__ kpu, float kscale, float sigma,
                            float* __restrict__ wf, int Nq, int Cin)
{
    extern __shared__ _Float16 smem[];
    int lane = threadIdx.x;
    int wave = threadIdx.y;
    int n = blockIdx.x * blockDim.y + wave;
    _Float16* aT = smem + (size_t)wave * (512 + 32 * (size_t)Cin);
    _Float16* fT = aT + 512;
    bool valid = (n < Nq);                  // wave-uniform
    if (valid) {
        int ni = neigh[(long)n * 32 + lane];
        float dx = sp[(long)ni * 3 + 0] - qp[(long)n * 3 + 0];
        float dy = sp[(long)ni * 3 + 1] - qp[(long)n * 3 + 1];
        float dz = sp[(long)ni * 3 + 2] - qp[(long)n * 3 + 2];
#pragma unroll
        for (int p = 0; p < 15; ++p) {
            float ex = dx - kpu[p * 3 + 0] * kscale;
            float ey = dy - kpu[p * 3 + 1] * kscale;
            float ez = dz - kpu[p * 3 + 2] * kscale;
            float d = sqrtf(ex * ex + ey * ey + ez * ez + 1e-12f);
            aT[p * 32 + lane] = (_Float16)fmaxf(0.f, 1.f - d / sigma);
        }
        aT[15 * 32 + lane] = (_Float16)0.f;   // pad row P=15
        const float* frow = x + (long)ni * Cin;
        for (int c = 0; c < Cin; c += 4) {
            float4 fv = *(const float4*)(frow + c);
            fT[lane * Cin + c + 0] = (_Float16)fv.x;
            fT[lane * Cin + c + 1] = (_Float16)fv.y;
            fT[lane * Cin + c + 2] = (_Float16)fv.z;
            fT[lane * Cin + c + 3] = (_Float16)fv.w;
        }
    }
    __syncthreads();
    if (!valid) return;
    int half = lane >> 4, lm = lane & 15;
    v16h a;
#pragma unroll
    for (int e = 0; e < 16; ++e) {
        int k = (e < 8) ? (half * 8 + e) : (16 + half * 8 + (e - 8));
        a[e] = aT[lm * 32 + k];
    }
    for (int t = 0; t < (Cin >> 4); ++t) {
        v16h b;
#pragma unroll
        for (int e = 0; e < 16; ++e)
            b[e] = fT[((e < 8) ? (half * 8 + e) : (16 + half * 8 + (e - 8))) * Cin + t * 16 + lm];
        v8f acc = {};
        acc = __builtin_amdgcn_wmma_f32_16x16x32_f16(false, a, false, b,
                                                     (short)0, acc, false, false);
#pragma unroll
        for (int r = 0; r < 8; ++r)
            wf[((long)n * 16 + half * 8 + r) * Cin + t * 16 + lm] = acc[r];
    }
}

// e11 special case: Cin == 1, scalar per point. Output padded to (Nq x 32)
// (cols 15..31 zero) so the e11 GEMM runs with K = 32 exactly.
__global__ void k_kpconv_wf_c1(const float* qp, const float* sp, const int* neigh,
                               const float* f1, const float* kpu, float kscale,
                               float sigma, float* wf32, int Nq)
{
    int n = blockIdx.x * blockDim.x + threadIdx.x;
    if (n >= Nq) return;
    float acc[15];
#pragma unroll
    for (int p = 0; p < 15; ++p) acc[p] = 0.f;
    float qx = qp[n * 3], qy = qp[n * 3 + 1], qz = qp[n * 3 + 2];
    for (int k = 0; k < 32; ++k) {
        int ni = neigh[n * 32 + k];
        float dx = sp[ni * 3] - qx, dy = sp[ni * 3 + 1] - qy, dz = sp[ni * 3 + 2] - qz;
        float fv = f1[ni];
#pragma unroll
        for (int p = 0; p < 15; ++p) {
            float ex = dx - kpu[p * 3] * kscale;
            float ey = dy - kpu[p * 3 + 1] * kscale;
            float ez = dz - kpu[p * 3 + 2] * kscale;
            float d = sqrtf(ex * ex + ey * ey + ez * ez + 1e-12f);
            acc[p] += fmaxf(0.f, 1.f - d / sigma) * fv;
        }
    }
    for (int p = 0; p < 15; ++p) wf32[(long)n * 32 + p] = acc[p];
    for (int p = 15; p < 32; ++p) wf32[(long)n * 32 + p] = 0.f;
}

// ---------------------------------------------------------------------------
// GroupNorm over (channels-in-group, all rows), 32 groups, optional LeakyReLU,
// in place. One block per group.
// ---------------------------------------------------------------------------
__global__ void k_groupnorm(float* __restrict__ x, const float* __restrict__ gs,
                            const float* __restrict__ gb, int N, int C, int do_lrelu)
{
    __shared__ float ssum[256], ssq[256];
    int g = blockIdx.x;
    int cpg = C / 32;
    long cnt = (long)N * cpg;
    float s = 0.f, q = 0.f;
    for (long i = threadIdx.x; i < cnt; i += 256) {
        int r = (int)(i / cpg);
        int c = g * cpg + (int)(i % cpg);
        float v = x[(long)r * C + c];
        s += v; q += v * v;
    }
    ssum[threadIdx.x] = s; ssq[threadIdx.x] = q;
    __syncthreads();
    for (int off = 128; off > 0; off >>= 1) {
        if ((int)threadIdx.x < off) {
            ssum[threadIdx.x] += ssum[threadIdx.x + off];
            ssq[threadIdx.x]  += ssq[threadIdx.x + off];
        }
        __syncthreads();
    }
    float mean = ssum[0] / (float)cnt;
    float var  = ssq[0] / (float)cnt - mean * mean;
    float inv  = rsqrtf(var + 1e-5f);
    for (long i = threadIdx.x; i < cnt; i += 256) {
        int r = (int)(i / cpg);
        int c = g * cpg + (int)(i % cpg);
        float v = (x[(long)r * C + c] - mean) * inv * gs[c] + gb[c];
        if (do_lrelu) v = lrelu_f(v);
        x[(long)r * C + c] = v;
    }
}

__global__ void k_add_lrelu(const float* a, const float* b, float* o, long n)
{
    long i = (long)blockIdx.x * blockDim.x + threadIdx.x;
    if (i < n) o[i] = lrelu_f(a[i] + b[i]);
}

__global__ void k_gather_max(const float* f, const int* neigh, float* o, int Nq, int C)
{
    long i = (long)blockIdx.x * blockDim.x + threadIdx.x;
    if (i >= (long)Nq * C) return;
    int n = (int)(i / C), c = (int)(i % C);
    const int* nb = neigh + (long)n * 32;
    float m = -3.4e38f;
    for (int k = 0; k < 32; ++k) m = fmaxf(m, f[(long)nb[k] * C + c]);
    o[i] = m;
}

__global__ void k_concat_gather(const float* A, const int* idx, int Ca,
                                const float* B, int Cb, float* o, int N)
{
    int Ct = Ca + Cb;
    long i = (long)blockIdx.x * blockDim.x + threadIdx.x;
    if (i >= (long)N * Ct) return;
    int n = (int)(i / Ct), c = (int)(i % Ct);
    o[i] = (c < Ca) ? A[(long)idx[n] * Ca + c] : B[(long)n * Cb + (c - Ca)];
}

// ---------------------------------------------------------------------------
// EGM: exact kNN(4) scan, attention/std, exact rank-based top-M, output.
// ---------------------------------------------------------------------------
__global__ void k_knn4(const float* pts, int N, int* out)
{
    int n = blockIdx.x * blockDim.x + threadIdx.x;
    if (n >= N) return;
    float px = pts[n * 3], py = pts[n * 3 + 1], pz = pts[n * 3 + 2];
    float bd[4] = {3.4e38f, 3.4e38f, 3.4e38f, 3.4e38f};
    int   bi[4] = {0, 0, 0, 0};
    for (int j = 0; j < N; ++j) {
        float dx = pts[j * 3] - px, dy = pts[j * 3 + 1] - py, dz = pts[j * 3 + 2] - pz;
        float d = dx * dx + dy * dy + dz * dz;
        if (d < bd[3]) {                // strict <: earliest index wins ties
            int p = 3;
            while (p > 0 && d < bd[p - 1]) { bd[p] = bd[p - 1]; bi[p] = bi[p - 1]; --p; }
            bd[p] = d; bi[p] = j;
        }
    }
    out[n * 4] = bi[0]; out[n * 4 + 1] = bi[1];
    out[n * 4 + 2] = bi[2]; out[n * 4 + 3] = bi[3];
}

__global__ void k_egm_att(const float* pts, const int* knn, const float* qw,
                          const float* kw, const float* vw,
                          float* att, float* vout, float* sel, int N)
{
    int n = blockIdx.x * blockDim.x + threadIdx.x;
    if (n >= N) return;
    float px = pts[n * 3], py = pts[n * 3 + 1], pz = pts[n * 3 + 2];
    float q0 = qw[0] * px + qw[1] * py + qw[2] * pz;
    float q1 = qw[3] * px + qw[4] * py + qw[5] * pz;
    float q2 = qw[6] * px + qw[7] * py + qw[8] * pz;
    const float invs = 0.57735026918962576f;   // 1/sqrt(3)
    float lg[4];
    for (int kk = 0; kk < 4; ++kk) {
        int j = knn[n * 4 + kk];
        float dx = pts[j * 3] - px, dy = pts[j * 3 + 1] - py, dz = pts[j * 3 + 2] - pz;
        float k0 = kw[0] * dx + kw[1] * dy + kw[2] * dz;
        float k1 = kw[3] * dx + kw[4] * dy + kw[5] * dz;
        float k2 = kw[6] * dx + kw[7] * dy + kw[8] * dz;
        lg[kk] = (q0 * k0 + q1 * k1 + q2 * k2) * invs;
        vout[(long)n * 12 + kk * 3 + 0] = vw[0] * dx + vw[1] * dy + vw[2] * dz;
        vout[(long)n * 12 + kk * 3 + 1] = vw[3] * dx + vw[4] * dy + vw[5] * dz;
        vout[(long)n * 12 + kk * 3 + 2] = vw[6] * dx + vw[7] * dy + vw[8] * dz;
    }
    float m = fmaxf(fmaxf(lg[0], lg[1]), fmaxf(lg[2], lg[3]));
    float e0 = expf(lg[0] - m), e1 = expf(lg[1] - m), e2 = expf(lg[2] - m), e3 = expf(lg[3] - m);
    float sm = e0 + e1 + e2 + e3;
    float a0 = e0 / sm, a1 = e1 / sm, a2 = e2 / sm, a3 = e3 / sm;
    att[n * 4] = a0; att[n * 4 + 1] = a1; att[n * 4 + 2] = a2; att[n * 4 + 3] = a3;
    float mean = 0.25f * (a0 + a1 + a2 + a3);
    float var  = 0.25f * (a0 * a0 + a1 * a1 + a2 * a2 + a3 * a3) - mean * mean;
    sel[n] = sqrtf(fmaxf(var, 0.f));
}

// Exact top-M by rank (descending value, ties -> lower index first).
__global__ void k_topm(const float* sel, int N, int M, int* top)
{
    int i = blockIdx.x * blockDim.x + threadIdx.x;
    if (i >= N) return;
    float s = sel[i];
    int r = 0;
    for (int j = 0; j < N; ++j) {
        float t = sel[j];
        r += (t > s) || (t == s && j < i);
    }
    if (r < M) top[r] = i;
}

__global__ void k_egm_out(const int* top, const float* att, const float* v,
                          const float* lw, const float* lb, float* pout, int M)
{
    int m = blockIdx.x * blockDim.x + threadIdx.x;
    if (m >= M) return;
    int i = top[m];
    float o0 = 0.f, o1 = 0.f, o2 = 0.f;
    for (int kk = 0; kk < 4; ++kk) {
        float a = att[i * 4 + kk];
        o0 += a * v[(long)i * 12 + kk * 3 + 0];
        o1 += a * v[(long)i * 12 + kk * 3 + 1];
        o2 += a * v[(long)i * 12 + kk * 3 + 2];
    }
    pout[m * 3 + 0] = lw[0] * o0 + lw[1] * o1 + lw[2] * o2 + lb[0];
    pout[m * 3 + 1] = lw[3] * o0 + lw[4] * o1 + lw[5] * o2 + lb[1];
    pout[m * 3 + 2] = lw[6] * o0 + lw[7] * o1 + lw[8] * o2 + lb[2];
}

__global__ void k_lin3(const float* x, const float* w, const float* b, float* y, int N)
{
    int n = blockIdx.x * blockDim.x + threadIdx.x;
    if (n >= N) return;
    float x0 = x[n * 3], x1 = x[n * 3 + 1], x2 = x[n * 3 + 2];
    float y0 = w[0] * x0 + w[1] * x1 + w[2] * x2 + b[0];
    float y1 = w[3] * x0 + w[4] * x1 + w[5] * x2 + b[1];
    float y2 = w[6] * x0 + w[7] * x1 + w[8] * x2 + b[2];
    y[n * 3] = y0; y[n * 3 + 1] = y1; y[n * 3 + 2] = y2;
}

// Exact numpy RandomState(7).uniform(-1,1,(15,3)) via MT19937 (one thread).
__global__ void k_init_kp(float* kp)
{
    if (threadIdx.x != 0 || blockIdx.x != 0) return;
    unsigned mt[624];
    mt[0] = 7u;
    for (int i = 1; i < 624; ++i)
        mt[i] = 1812433253u * (mt[i - 1] ^ (mt[i - 1] >> 30)) + (unsigned)i;
    for (int i = 0; i < 624; ++i) {      // one in-place twist (need only 90 draws)
        unsigned y = (mt[i] & 0x80000000u) | (mt[(i + 1) % 624] & 0x7fffffffu);
        mt[i] = mt[(i + 397) % 624] ^ (y >> 1) ^ ((y & 1u) ? 2567483615u : 0u);
    }
    int c = 0;
    for (int i = 0; i < 45; ++i) {
        unsigned y0 = mt[c++];
        y0 ^= y0 >> 11; y0 ^= (y0 << 7) & 2636928640u; y0 ^= (y0 << 15) & 4022730752u; y0 ^= y0 >> 18;
        unsigned y1 = mt[c++];
        y1 ^= y1 >> 11; y1 ^= (y1 << 7) & 2636928640u; y1 ^= (y1 << 15) & 4022730752u; y1 ^= y1 >> 18;
        double u = ((double)(y0 >> 5) * 67108864.0 + (double)(y1 >> 6)) / 9007199254740992.0;
        kp[i] = (float)(-1.0 + 2.0 * u);
    }
}

// ---------------------------------------------------------------------------
// Host-side orchestration
// ---------------------------------------------------------------------------
static inline unsigned gdim(long n, int b) { return (unsigned)((n + b - 1) / b); }

static void gemm(hipStream_t s, const float* A, const float* B, const float* bias,
                 float* C, int M, int N, int K)
{
    dim3 blk(32, 4, 1);
    dim3 grd((unsigned)(N / 16), (unsigned)((M + 63) / 64), 1);
    k_gemm_bias<<<grd, blk, 0, s>>>(A, B, bias, C, M, N, K);
}

static void kpconv_wf(hipStream_t s, const float* qp, const float* sp, const int* neigh,
                      const float* x, const float* kp, float kscale, float sigma,
                      float* wf, int Nq, int Cin)
{
    dim3 blk(32, 2, 1);
    dim3 grd(gdim(Nq, 2), 1, 1);
    size_t sh = (size_t)2 * (512 + 32 * (size_t)Cin) * sizeof(_Float16);
    k_kpconv_wf<<<grd, blk, sh, s>>>(qp, sp, neigh, x, kp, kscale, sigma, wf, Nq, Cin);
}

struct ResP {
    const float *u1w, *u1b, *u1gs, *u1gb, *kw, *g2s, *g2b, *u2w, *u2b, *g3s, *g3b;
    const float *sw, *sb, *sgs, *sgb;
};

static ResP res_from(void* const* in, int base, bool has_sw)
{
    ResP r;
    r.u1w = (const float*)in[base + 0];  r.u1b = (const float*)in[base + 1];
    r.u1gs = (const float*)in[base + 2]; r.u1gb = (const float*)in[base + 3];
    r.kw = (const float*)in[base + 4];
    r.g2s = (const float*)in[base + 5];  r.g2b = (const float*)in[base + 6];
    r.u2w = (const float*)in[base + 7];  r.u2b = (const float*)in[base + 8];
    r.g3s = (const float*)in[base + 9];  r.g3b = (const float*)in[base + 10];
    if (has_sw) {
        r.sw = (const float*)in[base + 11]; r.sb = (const float*)in[base + 12];
        r.sgs = (const float*)in[base + 13]; r.sgb = (const float*)in[base + 14];
    } else { r.sw = nullptr; r.sb = nullptr; r.sgs = nullptr; r.sgb = nullptr; }
    return r;
}

static void run_res(hipStream_t s, const float* F, int Ns, int Cin,
                    const float* qp, int Nq, const float* sp, const int* neigh,
                    const ResP& P, int Cout, const float* kp, float kscale, float sigma,
                    bool strided, float* O, float* T1, float* T2, float* SC,
                    float* WF, float* BP)
{
    int h = Cout / 4;
    gemm(s, F, P.u1w, P.u1b, T1, Ns, h, Cin);                       // unary1
    k_groupnorm<<<32, 256, 0, s>>>(T1, P.u1gs, P.u1gb, Ns, h, 1);
    kpconv_wf(s, qp, sp, neigh, T1, kp, kscale, sigma, WF, Nq, h);  // wf (WMMA)
    // kpconv weights (15h x h) -> padded weight layout (h x 16h)
    k_pad_transpose<<<gdim((long)h * 16 * h, 256), 256, 0, s>>>(P.kw, BP, h, 15 * h, 16 * h);
    gemm(s, WF, BP, nullptr, T2, Nq, h, 16 * h);                    // conv GEMM
    k_groupnorm<<<32, 256, 0, s>>>(T2, P.g2s, P.g2b, Nq, h, 1);
    gemm(s, T2, P.u2w, P.u2b, T1, Nq, Cout, h);                     // unary2
    k_groupnorm<<<32, 256, 0, s>>>(T1, P.g3s, P.g3b, Nq, Cout, 0);
    const float* scsrc = F;
    if (strided) {
        long tot = (long)Nq * Cin;
        k_gather_max<<<gdim(tot, 256), 256, 0, s>>>(F, neigh, SC, Nq, Cin);
        scsrc = SC;
    }
    const float* scf = scsrc;
    if (P.sw) {
        gemm(s, scsrc, P.sw, P.sb, T2, Nq, Cout, Cin);
        k_groupnorm<<<32, 256, 0, s>>>(T2, P.sgs, P.sgb, Nq, Cout, 0);
        scf = T2;
    }
    long tot = (long)Nq * Cout;
    k_add_lrelu<<<gdim(tot, 256), 256, 0, s>>>(T1, scf, O, tot);
}

static void run_egm(hipStream_t s, const float* pts, int N, int M,
                    const float* qw, const float* kw, const float* vw,
                    const float* lw, const float* lb, float* pout,
                    float* att, float* v, float* sel, int* knn, int* top)
{
    k_knn4<<<gdim(N, 128), 128, 0, s>>>(pts, N, knn);
    k_egm_att<<<gdim(N, 128), 128, 0, s>>>(pts, knn, qw, kw, vw, att, v, sel, N);
    k_topm<<<gdim(N, 128), 128, 0, s>>>(sel, N, M, top);
    k_egm_out<<<gdim(M, 128), 128, 0, s>>>(top, att, v, lw, lb, pout, M);
}

// Workspace arena (float element offsets); ~37 MB total.
static const long OFF_F1  = 0;                       // 8192x128
static const long OFF_T1  = OFF_F1  + 1048576;       // scratch (<=1M)
static const long OFF_T2  = OFF_T1  + 1048576;       // scratch (<=1M)
static const long OFF_SC  = OFF_T2  + 1048576;       // gather-max (<=2048x128)
static const long OFF_WF  = OFF_SC  + 262144;        // wf padded (<=8192x16x32)
static const long OFF_BP  = OFF_WF  + 4194304;       // padded conv W (<=256x4096)
static const long OFF_F2  = OFF_BP  + 1048576;       // 2048x128
static const long OFF_F3  = OFF_F2  + 262144;        // 512x128
static const long OFF_F4A = OFF_F3  + 65536;         // 128x256 (e41 out)
static const long OFF_P0N = OFF_F4A + 32768;         // 8192x3
static const long OFF_P1  = OFF_P0N + 24576;         // 2048x3
static const long OFF_P2  = OFF_P1  + 6144;          // 512x3
static const long OFF_P3  = OFF_P2  + 1536;          // 128x3
static const long OFF_KP  = OFF_P3  + 384;           // 45 (KP_UNIT)
static const long OFF_ATT = OFF_KP  + 64;            // 8192x4
static const long OFF_V   = OFF_ATT + 32768;         // 8192x4x3
static const long OFF_SEL = OFF_V   + 98304;         // 8192
static const long OFF_KNN = OFF_SEL + 8192;          // 8192x4 (int)
static const long OFF_TOP = OFF_KNN + 32768;         // 2048 (int)

extern "C" void kernel_launch(void* const* d_in, const int* in_sizes, int n_in,
                              void* d_out, int out_size, void* d_ws, size_t ws_size,
                              hipStream_t stream)
{
    (void)in_sizes; (void)n_in; (void)out_size; (void)ws_size;
    const float R0 = 0.1f, S0 = 0.1f;

    const float* feats = (const float*)d_in[0];
    const float* p0    = (const float*)d_in[1];
    const int* n0 = (const int*)d_in[2];
    const int* n1 = (const int*)d_in[3];
    const int* n2 = (const int*)d_in[4];
    const int* n3 = (const int*)d_in[5];
    const int* s0 = (const int*)d_in[6];
    const int* s1 = (const int*)d_in[7];
    const int* s2 = (const int*)d_in[8];
    const int* up1 = (const int*)d_in[9];
    const int* up2 = (const int*)d_in[10];

    const float* e11kw = (const float*)d_in[11];
    const float* e11gs = (const float*)d_in[12];
    const float* e11gb = (const float*)d_in[13];
    ResP E12 = res_from(d_in, 14, true);
    ResP E21 = res_from(d_in, 29, false);
    ResP E22 = res_from(d_in, 40, false);
    ResP E31 = res_from(d_in, 51, false);
    ResP E32 = res_from(d_in, 62, false);
    ResP E41 = res_from(d_in, 73, true);
    ResP E42 = res_from(d_in, 88, true);
    const float* g1q = (const float*)d_in[103]; const float* g1k = (const float*)d_in[104];
    const float* g1v = (const float*)d_in[105];
    const float* g2q = (const float*)d_in[106]; const float* g2k = (const float*)d_in[107];
    const float* g2v = (const float*)d_in[108];
    const float* g3q = (const float*)d_in[109]; const float* g3k = (const float*)d_in[110];
    const float* g3v = (const float*)d_in[111];
    const float* l1w = (const float*)d_in[112]; const float* l1b = (const float*)d_in[113];
    const float* l2w = (const float*)d_in[114]; const float* l2b = (const float*)d_in[115];
    const float* l3w = (const float*)d_in[116]; const float* l3b = (const float*)d_in[117];
    const float* d3uw = (const float*)d_in[118]; const float* d3ub = (const float*)d_in[119];
    const float* d3gs = (const float*)d_in[120]; const float* d3gb = (const float*)d_in[121];
    const float* d2w = (const float*)d_in[122]; const float* d2b = (const float*)d_in[123];

    float* W = (float*)d_ws;
    float* F1 = W + OFF_F1;  float* T1 = W + OFF_T1;  float* T2 = W + OFF_T2;
    float* SC = W + OFF_SC;  float* WF = W + OFF_WF;  float* BP = W + OFF_BP;
    float* F2 = W + OFF_F2;  float* F3 = W + OFF_F3;  float* F4A = W + OFF_F4A;
    float* P0N = W + OFF_P0N; float* P1 = W + OFF_P1; float* P2 = W + OFF_P2;
    float* P3 = W + OFF_P3;  float* KP = W + OFF_KP;
    float* ATT = W + OFF_ATT; float* Vb = W + OFF_V;  float* SEL = W + OFF_SEL;
    int* KNN = (int*)(W + OFF_KNN);
    int* TOP = (int*)(W + OFF_TOP);

    float* out_l2 = (float*)d_out;                 // 2048x64
    float* out_l3 = out_l2 + 2048 * 64;            // 512x128
    float* out_f4 = out_l3 + 512 * 128;            // 128x1024

    hipStream_t s = stream;

    k_init_kp<<<1, 1, 0, s>>>(KP);                 // KP_UNIT (MT19937 seed 7)

    // stage 1: e11 (Cin=1) + GN + lrelu; wf padded to (8192 x 32), K=32 exact
    k_kpconv_wf_c1<<<gdim(8192, 128), 128, 0, s>>>(p0, p0, n0, feats, KP, R0, S0, WF, 8192);
    k_pad_transpose<<<gdim(64 * 32, 256), 256, 0, s>>>(e11kw, BP, 64, 15, 32);
    gemm(s, WF, BP, nullptr, F1, 8192, 64, 32);
    k_groupnorm<<<32, 256, 0, s>>>(F1, e11gs, e11gb, 8192, 64, 1);
    // e12 (64 -> 128)
    run_res(s, F1, 8192, 64, p0, 8192, p0, n0, E12, 128, KP, R0, S0, false,
            F1, T1, T2, SC, WF, BP);

    // EGM 0->1 and point transforms
    run_egm(s, p0, 8192, 2048, g1q, g1k, g1v, l1w, l1b, P1, ATT, Vb, SEL, KNN, TOP);
    k_lin3<<<gdim(8192, 128), 128, 0, s>>>(p0, l1w, l1b, P0N, 8192);

    run_res(s, F1, 8192, 128, P1, 2048, P0N, s0, E21, 128, KP, R0, S0, true,
            F2, T1, T2, SC, WF, BP);
    run_res(s, F2, 2048, 128, P1, 2048, P1, n1, E22, 128, KP, 2*R0, 2*S0, false,
            F2, T1, T2, SC, WF, BP);

    // EGM 1->2
    run_egm(s, P1, 2048, 512, g2q, g2k, g2v, l2w, l2b, P2, ATT, Vb, SEL, KNN, TOP);
    k_lin3<<<gdim(2048, 128), 128, 0, s>>>(P1, l2w, l2b, P1, 2048);

    run_res(s, F2, 2048, 128, P2, 512, P1, s1, E31, 128, KP, 2*R0, 2*S0, true,
            F3, T1, T2, SC, WF, BP);
    run_res(s, F3, 512, 128, P2, 512, P2, n2, E32, 128, KP, 4*R0, 4*S0, false,
            F3, T1, T2, SC, WF, BP);

    // EGM 2->3
    run_egm(s, P2, 512, 128, g3q, g3k, g3v, l3w, l3b, P3, ATT, Vb, SEL, KNN, TOP);
    k_lin3<<<gdim(512, 128), 128, 0, s>>>(P2, l3w, l3b, P2, 512);

    run_res(s, F3, 512, 128, P3, 128, P2, s2, E41, 256, KP, 4*R0, 4*S0, true,
            F4A, T1, T2, SC, WF, BP);
    run_res(s, F4A, 128, 256, P3, 128, P3, n3, E42, 1024, KP, 8*R0, 8*S0, false,
            out_f4, T1, T2, SC, WF, BP);

    // decoder
    k_concat_gather<<<gdim((long)512 * 1152, 256), 256, 0, s>>>(out_f4, up2, 1024, F3, 128, T1, 512);
    gemm(s, T1, d3uw, d3ub, out_l3, 512, 128, 1152);
    k_groupnorm<<<32, 256, 0, s>>>(out_l3, d3gs, d3gb, 512, 128, 1);
    k_concat_gather<<<gdim((long)2048 * 256, 256), 256, 0, s>>>(out_l3, up1, 128, F2, 128, T1, 2048);
    gemm(s, T1, d2w, d2b, out_l2, 2048, 64, 256);
}
